// InferencePalettizedLinear_44324062494774
// MI455X (gfx1250) — compile-verified, matrix operand order checked by reading, not run on hardware
//
#include <hip/hip_runtime.h>

// Problem sizes (fixed by the reference)
#define IN_F    4096
#define OUT_F   16384
#define M_TOTAL 8192            // 4 * 2048 rows
#define PALETTE 256

// GEMM tiling
#define BM  128
#define BN  128
#define BK  64
#define LDK (BK + 8)            // 72 elems = 144 B pitch: 16B-aligned, bank-conflict-free b128
#define NK  (IN_F / BK)         // 64 K-iterations

typedef __attribute__((ext_vector_type(16))) __bf16 v16bf;
typedef __attribute__((ext_vector_type(8)))  float  v8f;

union FragU { v16bf v; uint4 q[2]; };

__device__ __forceinline__ unsigned short f32_to_bf16(float f) {
    unsigned int u = __float_as_uint(f);
    u += 0x7FFFu + ((u >> 16) & 1u);      // round to nearest even
    return (unsigned short)(u >> 16);
}

// CDNA5 async global->LDS copy (16B), tracked by ASYNCcnt. vdst = LDS byte
// address (low 32 bits of generic shared pointer), vaddr = 64-bit global addr.
__device__ __forceinline__ void async_copy_b128(const void* gaddr, void* lds) {
    unsigned lds_addr = (unsigned)(unsigned long long)lds;
    asm volatile("global_load_async_to_lds_b128 %0, %1, off"
                 :
                 : "v"(lds_addr), "v"(gaddr)
                 : "memory");
}

// ---------------------------------------------------------------------------
// Kernel 1: dequantize palette weights: int32 idx -> bf16 bits, row-major
// ---------------------------------------------------------------------------
__global__ __launch_bounds__(256) void dequant_w(const int* __restrict__ idx,
                                                 const float* __restrict__ lut,
                                                 unsigned short* __restrict__ w) {
    __shared__ float slut[PALETTE];
    slut[threadIdx.x] = lut[threadIdx.x];   // blockDim.x == 256 == PALETTE
    __syncthreads();

    size_t base = ((size_t)blockIdx.x * 256 + threadIdx.x) * 8;
    int4 i0 = *(const int4*)(idx + base);
    int4 i1 = *(const int4*)(idx + base + 4);
    uint4 o;
    o.x = (unsigned)f32_to_bf16(slut[i0.x]) | ((unsigned)f32_to_bf16(slut[i0.y]) << 16);
    o.y = (unsigned)f32_to_bf16(slut[i0.z]) | ((unsigned)f32_to_bf16(slut[i0.w]) << 16);
    o.z = (unsigned)f32_to_bf16(slut[i1.x]) | ((unsigned)f32_to_bf16(slut[i1.y]) << 16);
    o.w = (unsigned)f32_to_bf16(slut[i1.z]) | ((unsigned)f32_to_bf16(slut[i1.w]) << 16);
    *(uint4*)(w + base) = o;
}

// ---------------------------------------------------------------------------
// Kernel 2: convert activations f32 -> bf16 bits, row-major
// ---------------------------------------------------------------------------
__global__ __launch_bounds__(256) void cvt_in(const float* __restrict__ x,
                                              unsigned short* __restrict__ y) {
    size_t base = ((size_t)blockIdx.x * 256 + threadIdx.x) * 8;
    float4 f0 = *(const float4*)(x + base);
    float4 f1 = *(const float4*)(x + base + 4);
    uint4 o;
    o.x = (unsigned)f32_to_bf16(f0.x) | ((unsigned)f32_to_bf16(f0.y) << 16);
    o.y = (unsigned)f32_to_bf16(f0.z) | ((unsigned)f32_to_bf16(f0.w) << 16);
    o.z = (unsigned)f32_to_bf16(f1.x) | ((unsigned)f32_to_bf16(f1.y) << 16);
    o.w = (unsigned)f32_to_bf16(f1.z) | ((unsigned)f32_to_bf16(f1.w) << 16);
    *(uint4*)(y + base) = o;
}

// ---------------------------------------------------------------------------
// Kernel 3: bf16 WMMA GEMM with double-buffered async global->LDS staging.
// Workgroup tile 128x128, BK=64; 8 wave32 waves as 2(M) x 4(N);
// each wave: 64x32 sub-tile = 4x2 accumulators of 16x16.
// ---------------------------------------------------------------------------
__global__ __launch_bounds__(256) void gemm_bf16(const unsigned short* __restrict__ A,
                                                 const unsigned short* __restrict__ W,
                                                 const float* __restrict__ bias,
                                                 float* __restrict__ out) {
    __shared__ unsigned short sA[2][BM][LDK];
    __shared__ unsigned short sB[2][BN][LDK];

    const int tid  = threadIdx.x;
    const int lane = tid & 31;
    const int wave = tid >> 5;      // 0..7
    const int wm   = wave & 1;      // wave row (2 along M)
    const int wn   = wave >> 1;     // wave col (4 along N)
    const int hi   = lane >> 4;     // half-wave selects K-subchunk (ISA 16-bit A/B layout)
    const int r    = lane & 15;     // row-within-tile for fragment loads

    const int mBase = blockIdx.y * BM;
    const int nBase = blockIdx.x * BN;

    // This thread's 4 staging chunks (row, col) — same for A and B tiles.
    int srow[4], scol[4];
    #pragma unroll
    for (int it = 0; it < 4; ++it) {
        int chunk = it * 256 + tid;        // 0..1023, 8 x 16B chunks per 64-elem row
        srow[it] = chunk >> 3;
        scol[it] = (chunk & 7) << 3;
    }

    v8f acc[4][2];
    #pragma unroll
    for (int i = 0; i < 4; ++i)
        #pragma unroll
        for (int j = 0; j < 2; ++j)
            #pragma unroll
            for (int e = 0; e < 8; ++e)
                acc[i][j][e] = 0.0f;

    // Prologue: stage tile 0 into buffer 0 (8 async b128 per thread).
    #pragma unroll
    for (int it = 0; it < 4; ++it) {
        async_copy_b128(A + (size_t)(mBase + srow[it]) * IN_F + scol[it],
                        &sA[0][srow[it]][scol[it]]);
        async_copy_b128(W + (size_t)(nBase + srow[it]) * IN_F + scol[it],
                        &sB[0][srow[it]][scol[it]]);
    }

    int cur = 0;
    for (int kt = 0; kt < NK; ++kt) {
        // Issue next tile's async loads into the other buffer, then wait only
        // for the *current* tile's (older) 8 loads to land.
        if (kt + 1 < NK) {
            const int k_next = (kt + 1) * BK;
            const int nxt = cur ^ 1;
            #pragma unroll
            for (int it = 0; it < 4; ++it) {
                async_copy_b128(A + (size_t)(mBase + srow[it]) * IN_F + k_next + scol[it],
                                &sA[nxt][srow[it]][scol[it]]);
                async_copy_b128(W + (size_t)(nBase + srow[it]) * IN_F + k_next + scol[it],
                                &sB[nxt][srow[it]][scol[it]]);
            }
            asm volatile("s_wait_asynccnt 0x8" ::: "memory");
        } else {
            asm volatile("s_wait_asynccnt 0x0" ::: "memory");
        }
        __syncthreads();   // current buffer fully resident for all waves

        #pragma unroll
        for (int ks = 0; ks < BK; ks += 32) {
            FragU a[4], b[2];
            // ISA 16-bit A layout: lane<16 holds row r, K = [h*8,h*8+8) U [16+h*8, ...)
            #pragma unroll
            for (int i = 0; i < 4; ++i) {
                int arow = wm * 64 + i * 16 + r;
                a[i].q[0] = *(const uint4*)&sA[cur][arow][ks + hi * 8];
                a[i].q[1] = *(const uint4*)&sA[cur][arow][ks + 16 + hi * 8];
            }
            // B column n = W row n (row-major W == column-major B): same pattern.
            #pragma unroll
            for (int j = 0; j < 2; ++j) {
                int brow = wn * 32 + j * 16 + r;
                b[j].q[0] = *(const uint4*)&sB[cur][brow][ks + hi * 8];
                b[j].q[1] = *(const uint4*)&sB[cur][brow][ks + 16 + hi * 8];
            }
            #pragma unroll
            for (int i = 0; i < 4; ++i)
                #pragma unroll
                for (int j = 0; j < 2; ++j)
                    acc[i][j] = __builtin_amdgcn_wmma_f32_16x16x32_bf16(
                        /*neg_a=*/false, a[i].v, /*neg_b=*/false, b[j].v,
                        /*c_mod=*/(short)0, acc[i][j],
                        /*reuse_a=*/false, /*reuse_b=*/false);
        }
        __syncthreads();   // all waves done reading `cur` before it is re-staged
        cur ^= 1;
    }

    // Epilogue: C/D layout -> lane<16: N=lane, M=vgpr; lane>=16: N=lane-16, M=vgpr+8.
    float bv[2];
    #pragma unroll
    for (int j = 0; j < 2; ++j)
        bv[j] = bias[nBase + wn * 32 + j * 16 + r];

    #pragma unroll
    for (int i = 0; i < 4; ++i) {
        #pragma unroll
        for (int j = 0; j < 2; ++j) {
            int col = nBase + wn * 32 + j * 16 + r;
            #pragma unroll
            for (int e = 0; e < 8; ++e) {
                int row = mBase + wm * 64 + i * 16 + hi * 8 + e;
                out[(size_t)row * OUT_F + col] = acc[i][j][e] + bv[j];
            }
        }
    }
}

// ---------------------------------------------------------------------------
extern "C" void kernel_launch(void* const* d_in, const int* in_sizes, int n_in,
                              void* d_out, int out_size, void* d_ws, size_t ws_size,
                              hipStream_t stream) {
    const float* inp  = (const float*)d_in[0];   // [4,2048,4096] f32
    const float* lut  = (const float*)d_in[1];   // [256] f32
    const int*   widx = (const int*)d_in[2];     // [16384,4096] i32
    const float* bias = (const float*)d_in[3];   // [16384] f32
    float*       out  = (float*)d_out;           // [4,2048,16384] f32

    // Workspace: 128 MB bf16 weights + 64 MB bf16 activations
    unsigned short* Wq = (unsigned short*)d_ws;
    unsigned short* Aq = Wq + (size_t)OUT_F * IN_F;

    {   // dequantize weights: 67.1M elems, 8 per thread
        size_t n = (size_t)OUT_F * IN_F;
        dequant_w<<<(unsigned)(n / (8 * 256)), 256, 0, stream>>>(widx, lut, Wq);
    }
    {   // convert activations: 33.5M elems, 8 per thread
        size_t n = (size_t)M_TOTAL * IN_F;
        cvt_in<<<(unsigned)(n / (8 * 256)), 256, 0, stream>>>(inp, Aq);
    }
    {   // GEMM
        dim3 grid(OUT_F / BN, M_TOTAL / BM);     // 128 x 64 workgroups
        gemm_bf16<<<grid, 256, 0, stream>>>(Aq, Wq, bias, out);
    }
}
